// NodePredictor_30236569764273
// MI455X (gfx1250) — compile-verified, matrix operand order checked by reading, not run on hardware
//
#include <hip/hip_runtime.h>
#include <hip/hip_bf16.h>

// ---------------------------------------------------------------------------
// GCN NodePredictor for MI455X (gfx1250, wave32).
//  - GEMMs via v_wmma_f32_16x16x32_bf16 (bf16 A/B, f32 accumulate)
//  - weights pre-packed per-call into the WMMA B-fragment lane layout
//  - ReLU specialized via template (no runtime blend); self-loop+bias
//    agg-init fused into GEMM epilogue; ReLU fused into FC read
//  - wave-per-edge coalesced scatter with f32 global atomics (L2-resident)
// ---------------------------------------------------------------------------

typedef __attribute__((ext_vector_type(16))) __bf16 v16bf;
typedef __attribute__((ext_vector_type(8)))  float  v8f;

#define HDIM 128
#define DEMB 64
#define NLAYER 6
#define OUTD 5
// packed bf16 weights per layer: 4 k-tiles * 8 n-tiles * 32 lanes * 16 elems
#define WPACK_PER_LAYER (4 * 8 * 32 * 16)

// ---------------- small utility kernels ----------------

__global__ void zero_f32_kernel(float* p, int n) {
    int i = blockIdx.x * blockDim.x + threadIdx.x;
    if (i < n) p[i] = 0.0f;
}

__global__ void degree_kernel(const int* __restrict__ ei, float* __restrict__ deg, int E) {
    int e = blockIdx.x * blockDim.x + threadIdx.x;
    if (e < E) atomicAdd(&deg[ei[E + e]], 1.0f);   // dst = ei[1][e]
}

__global__ void rsqrt_kernel(float* __restrict__ deg, int n) {
    int i = blockIdx.x * blockDim.x + threadIdx.x;
    if (i < n) deg[i] = rsqrtf(deg[i] + 1.0f);     // dis = (deg+1)^{-1/2}, in place
}

// x0[n,c] = emb{0,1}[x_cat[n,f]][cc] + PE(n,cc),  f = c/64, cc = c%64
__global__ void embed_pe_kernel(const int* __restrict__ x_cat,
                                const float* __restrict__ emb0,
                                const float* __restrict__ emb1,
                                float* __restrict__ x0, int N) {
    int t = blockIdx.x * blockDim.x + threadIdx.x;
    if (t >= N * HDIM) return;
    int n  = t >> 7;
    int c  = t & (HDIM - 1);
    int f  = c >> 6;
    int cc = c & (DEMB - 1);
    int tok = x_cat[n * 2 + f];
    const float* emb = f ? emb1 : emb0;
    float e = emb[(size_t)tok * DEMB + cc];
    // pe(n,cc): k2 = 2*(cc/2); ang = n * exp(-k2 * ln(10000)/64)
    float k2  = (float)(cc & ~1);
    float ang = (float)n * __expf(-k2 * (9.210340371976184f / 64.0f));
    float pe  = (cc & 1) ? __cosf(ang) : __sinf(ang);
    x0[t] = e + pe;
}

// Pack gcn_w [L,128,128] f32 -> bf16 B-fragments in WMMA lane layout.
// Within a layer, flat index = kt*4096 + nt*512 + lane*16 + i, holding
// W[k][n] with n = nt*16 + lane%16, kbase = kt*32 + (lane/16)*8,
// k = kbase+i (i<8) or kbase+16+(i-8) (i>=8)   [ISA 7.12.2, 16-bit B layout]
__global__ void pack_w_kernel(const float* __restrict__ W, unsigned short* __restrict__ packed) {
    int t = blockIdx.x * blockDim.x + threadIdx.x;
    if (t >= NLAYER * WPACK_PER_LAYER) return;
    int i    = t & 15;
    int lane = (t >> 4) & 31;
    int nt   = (t >> 9) & 7;
    int kt   = (t >> 12) & 3;
    int l    = t >> 14;
    int n     = nt * 16 + (lane & 15);
    int kbase = kt * 32 + (lane >> 4) * 8;
    int k     = (i < 8) ? (kbase + i) : (kbase + 16 + (i - 8));
    float w = W[((size_t)l * HDIM + k) * HDIM + n];
    __hip_bfloat16 b = __float2bfloat16(w);
    packed[t] = *reinterpret_cast<unsigned short*>(&b);
}

// ------- WMMA GEMM + fused agg-init:
//   h[N,128]   = (RELU?)(xin) @ Wl
//   agg[N,128] = h * dis^2 + bias          (self-loop term, pre-scatter)
// One wave computes a 16x128 output stripe: 8 N-tiles x 4 K-tiles of
// v_wmma_f32_16x16x32_bf16. A fragments built from f32 input; RELU is a
// compile-time specialization so no blend/select code is generated.
template <int RELU>
__global__ __launch_bounds__(256) void gemm_bf16_wmma_kernel(
    const float* __restrict__ xin, const unsigned short* __restrict__ wpack,
    const float* __restrict__ dis, const float* __restrict__ bias,
    float* __restrict__ hout, float* __restrict__ agg, int N) {
    int wave = threadIdx.x >> 5;
    int lane = threadIdx.x & 31;
    int tile = blockIdx.x * 8 + wave;
    int ntiles = (N + 15) >> 4;
    if (tile >= ntiles) return;

    int row0 = tile << 4;
    int m    = lane & 15;   // row within tile / col within C tile
    int half = lane >> 4;   // lane group selects K sub-block
    int row  = row0 + m;
    if (row >= N) row = N - 1;          // clamp (stores guarded below)
    const float* xr = xin + (size_t)row * HDIM;

    // Build 4 A fragments (16-bit A layout: lane holds row m; VGPR0..3 hold
    // K=kbase..kbase+7, VGPR4..7 hold K=kbase+16..kbase+23, kbase=half*8)
    v16bf a[4];
#pragma unroll
    for (int kt = 0; kt < 4; ++kt) {
        int kbase = kt * 32 + half * 8;
        float4 q0 = *(const float4*)(xr + kbase);
        float4 q1 = *(const float4*)(xr + kbase + 4);
        float4 q2 = *(const float4*)(xr + kbase + 16);
        float4 q3 = *(const float4*)(xr + kbase + 20);
        float v[16] = {q0.x, q0.y, q0.z, q0.w, q1.x, q1.y, q1.z, q1.w,
                       q2.x, q2.y, q2.z, q2.w, q3.x, q3.y, q3.z, q3.w};
#pragma unroll
        for (int j = 0; j < 16; ++j) {
            float t = v[j];
            if (RELU) t = fmaxf(t, 0.0f);
            a[kt][j] = (__bf16)t;
        }
    }

    // dis^2 for the 8 rows this lane's C fragment covers (r + 8*half)
    float d2v[8];
#pragma unroll
    for (int r = 0; r < 8; ++r) {
        int gr = row0 + r + half * 8;
        float d = dis[gr < N ? gr : (N - 1)];
        d2v[r] = d * d;
    }
    const bool full = (row0 + 16 <= N);   // wave-uniform: tile fully in range

#pragma unroll
    for (int nt = 0; nt < 8; ++nt) {
        v8f c = {0.f, 0.f, 0.f, 0.f, 0.f, 0.f, 0.f, 0.f};
#pragma unroll
        for (int kt = 0; kt < 4; ++kt) {
            // contiguous 32B fragment load per lane
            v16bf b = *(const v16bf*)(wpack + (((kt * 8 + nt) * 32 + lane) << 4));
            c = __builtin_amdgcn_wmma_f32_16x16x32_bf16(
                    /*neg_a=*/false, a[kt], /*neg_b=*/false, b,
                    /*c_mod=*/(short)0, c, /*reuse_a=*/false, /*reuse_b=*/false);
        }
        // C layout: VGPR r -> row (r + 8*half), col = m
        int col = nt * 16 + m;
        float bcol = bias[col];
        if (full) {
#pragma unroll
            for (int r = 0; r < 8; ++r) {
                size_t off = (size_t)(row0 + r + half * 8) * HDIM + col;
                hout[off] = c[r];
                agg[off]  = c[r] * d2v[r] + bcol;
            }
        } else {
#pragma unroll
            for (int r = 0; r < 8; ++r) {
                int gr = row0 + r + half * 8;
                if (gr < N) {
                    size_t off = (size_t)gr * HDIM + col;
                    hout[off] = c[r];
                    agg[off]  = c[r] * d2v[r] + bcol;
                }
            }
        }
    }
}

// one wave per edge: lane c covers channels 4c..4c+3 (coalesced 512B read of
// h[src], 4 f32 atomics into agg[dst]; norm computed inline)
__global__ __launch_bounds__(256) void scatter_kernel(
    const float* __restrict__ h, const int* __restrict__ ei,
    const float* __restrict__ dis, float* __restrict__ agg, int E) {
    int e = blockIdx.x * 8 + (threadIdx.x >> 5);
    if (e >= E) return;
    int lane = threadIdx.x & 31;
    int src = ei[e];
    int dst = ei[E + e];
    float nrm = dis[src] * dis[dst];
    float4 v = *(const float4*)(h + (size_t)src * HDIM + lane * 4);
    float* ap = agg + (size_t)dst * HDIM + lane * 4;
    atomicAdd(ap + 0, v.x * nrm);
    atomicAdd(ap + 1, v.y * nrm);
    atomicAdd(ap + 2, v.z * nrm);
    atomicAdd(ap + 3, v.w * nrm);
}

// out[n,:] = relu(agg[n,:]) @ fc_w + fc_b    (ReLU of last layer fused here)
__global__ void fc_kernel(const float* __restrict__ agg, const float* __restrict__ fw,
                          const float* __restrict__ fb, float* __restrict__ out, int N) {
    int n = blockIdx.x * blockDim.x + threadIdx.x;
    if (n >= N) return;
    float acc[OUTD];
#pragma unroll
    for (int o = 0; o < OUTD; ++o) acc[o] = fb[o];
    const float* r = agg + (size_t)n * HDIM;
    for (int c = 0; c < HDIM; ++c) {
        float v = fmaxf(r[c], 0.0f);
#pragma unroll
        for (int o = 0; o < OUTD; ++o) acc[o] += v * fw[c * OUTD + o];
    }
#pragma unroll
    for (int o = 0; o < OUTD; ++o) out[n * OUTD + o] = acc[o];
}

// ---------------- host-side orchestration ----------------

extern "C" void kernel_launch(void* const* d_in, const int* in_sizes, int n_in,
                              void* d_out, int out_size, void* d_ws, size_t ws_size,
                              hipStream_t stream) {
    const int*   x_cat = (const int*)d_in[0];
    const int*   ei    = (const int*)d_in[1];
    const float* emb0  = (const float*)d_in[2];
    const float* emb1  = (const float*)d_in[3];
    const float* gcn_w = (const float*)d_in[4];
    const float* gcn_b = (const float*)d_in[5];
    const float* fc_w  = (const float*)d_in[6];
    const float* fc_b  = (const float*)d_in[7];
    float* out = (float*)d_out;

    const int N = in_sizes[0] / 2;   // 50000
    const int E = in_sizes[1] / 2;   // 800000
    const size_t nh = (size_t)N * HDIM;

    // carve workspace (256B-aligned sub-buffers): x0, h, agg, dis, packed W
    char* w = (char*)d_ws;
    auto take = [&](size_t bytes) {
        char* p = w;
        w += (bytes + 255) & ~(size_t)255;
        return p;
    };
    float*          x0    = (float*)take(nh * sizeof(float));
    float*          hbuf  = (float*)take(nh * sizeof(float));
    float*          agg   = (float*)take(nh * sizeof(float));
    float*          dis   = (float*)take((size_t)N * sizeof(float));
    unsigned short* wpack = (unsigned short*)take((size_t)NLAYER * WPACK_PER_LAYER * sizeof(unsigned short));

    const int T = 256;
    // degrees -> dis (in place)
    zero_f32_kernel<<<(N + T - 1) / T, T, 0, stream>>>(dis, N);
    degree_kernel<<<(E + T - 1) / T, T, 0, stream>>>(ei, dis, E);
    rsqrt_kernel<<<(N + T - 1) / T, T, 0, stream>>>(dis, N);
    // embeddings + positional encoding
    embed_pe_kernel<<<(int)((nh + T - 1) / T), T, 0, stream>>>(x_cat, emb0, emb1, x0, N);
    // pack weights to bf16 WMMA fragments
    pack_w_kernel<<<(NLAYER * WPACK_PER_LAYER + T - 1) / T, T, 0, stream>>>(gcn_w, wpack);

    const int ntiles  = (N + 15) >> 4;
    const int gblocks = (ntiles + 7) >> 3;
    const float* cur = x0;
    for (int l = 0; l < NLAYER; ++l) {
        if (l == 0) {
            gemm_bf16_wmma_kernel<0><<<gblocks, T, 0, stream>>>(
                cur, wpack + (size_t)l * WPACK_PER_LAYER, dis, gcn_b + l * HDIM,
                hbuf, agg, N);
        } else {
            gemm_bf16_wmma_kernel<1><<<gblocks, T, 0, stream>>>(
                cur, wpack + (size_t)l * WPACK_PER_LAYER, dis, gcn_b + l * HDIM,
                hbuf, agg, N);
        }
        scatter_kernel<<<(E + 7) / 8, T, 0, stream>>>(hbuf, ei, dis, agg, E);
        cur = agg;
    }
    fc_kernel<<<(N + T - 1) / T, T, 0, stream>>>(agg, fc_w, fc_b, out, N);
}
